// ResTokEncoder_46789373722978
// MI455X (gfx1250) — compile-verified, matrix operand order checked by reading, not run on hardware
//
#include <hip/hip_runtime.h>
#include <hip/hip_bf16.h>
#include <math.h>

typedef __attribute__((ext_vector_type(16))) _Float16 v16h;
typedef __attribute__((ext_vector_type(8)))  _Float16 v8h;
typedef __attribute__((ext_vector_type(8)))  float    v8f;

union U16 { v16h v; v8h h[2]; };

#define NTOK 341
#define DMODEL 1024
#define NHEADS 16
#define HD 64
#define FFDIM 4096
#define MROWS 2728           // B*NTOK = 8*341
#define NEGBIG (-1e9f)

#if __has_builtin(__builtin_amdgcn_global_load_async_to_lds_b128) && \
    __has_builtin(__builtin_amdgcn_s_wait_asynccnt)
#define HAVE_ASYNC 1
#else
#define HAVE_ASYNC 0
#endif

#if HAVE_ASYNC
// The async builtin expects (v4i AS1*, v4i AS3*, Imm int, Imm int) where v4i is
// a GCC-vector int4. Build exactly-typed pointers via ptr->int->AS-ptr (generic
// LDS pointers keep the LDS offset in the low 32 bits; generic global pointer
// bits equal the global address).
typedef int v4i_gcc __attribute__((vector_size(16)));
typedef __attribute__((address_space(1))) v4i_gcc as1_v4i;
typedef __attribute__((address_space(3))) v4i_gcc as3_v4i;
__device__ __forceinline__ as1_v4i* to_global(const void* p) {
    return (as1_v4i*)(unsigned long long)(uintptr_t)p;
}
__device__ __forceinline__ as3_v4i* to_lds(void* p) {
    return (as3_v4i*)(unsigned)(uintptr_t)p;
}
#endif

// ---------------------------------------------------------------------------
// Multi-scale token pyramid with residual (levels 1,2,4,8,16 on a 16x16 grid)
// ---------------------------------------------------------------------------
__global__ void tokens_kernel(const float* __restrict__ x, float* __restrict__ out) {
    int bc = blockIdx.x;              // b*1024 + c
    int b  = bc >> 10, c = bc & 1023;
    int tid = threadIdx.x;            // 0..255
    int row = tid >> 4, col = tid & 15;
    __shared__ float buf[256];
    float rest = x[((size_t)b * DMODEL + c) * 256 + tid];

    const int soff[5]  = {0, 1, 5, 21, 85};
    const int svals[5] = {1, 2, 4, 8, 16};
    for (int li = 0; li < 5; ++li) {
        int s = svals[li];
        int f = 16 / s;
        buf[tid] = rest;
        __syncthreads();
        int br = row / f, bcx = col / f;
        float sum = 0.f;
        for (int i = 0; i < f; ++i)
            for (int j = 0; j < f; ++j)
                sum += buf[(br * f + i) * 16 + (bcx * f + j)];
        float pooled = sum / (float)(f * f);
        if ((row % f == 0) && (col % f == 0)) {
            int tok = soff[li] + br * s + bcx;
            out[((size_t)b * NTOK + tok) * DMODEL + c] = pooled;
        }
        rest -= pooled;
        __syncthreads();
    }
}

// ---------------------------------------------------------------------------
// LayerNorm over D=1024, one block (256 threads) per token
// ---------------------------------------------------------------------------
__global__ void ln_kernel(const float* __restrict__ in, const float* __restrict__ w,
                          const float* __restrict__ b, float* __restrict__ out) {
    int m = blockIdx.x;
    int tid = threadIdx.x;
    const float4 v = *(const float4*)(in + (size_t)m * DMODEL + tid * 4);
    float s  = v.x + v.y + v.z + v.w;
    float ss = v.x * v.x + v.y * v.y + v.z * v.z + v.w * v.w;
    for (int o = 1; o < 32; o <<= 1) {
        s  += __shfl_xor(s,  o, 32);
        ss += __shfl_xor(ss, o, 32);
    }
    __shared__ float sm[8], sq[8];
    int wid = tid >> 5, lane = tid & 31;
    if (lane == 0) { sm[wid] = s; sq[wid] = ss; }
    __syncthreads();
    if (tid == 0) {
        float a = 0.f, a2 = 0.f;
        for (int i = 0; i < 8; ++i) { a += sm[i]; a2 += sq[i]; }
        sm[0] = a; sq[0] = a2;
    }
    __syncthreads();
    float mean = sm[0] * (1.f / 1024.f);
    float var  = sq[0] * (1.f / 1024.f) - mean * mean;
    float rstd = rsqrtf(var + 1e-5f);
    const float4 w4 = *(const float4*)(w + tid * 4);
    const float4 b4 = *(const float4*)(b + tid * 4);
    float4 o4;
    o4.x = (v.x - mean) * rstd * w4.x + b4.x;
    o4.y = (v.y - mean) * rstd * w4.y + b4.y;
    o4.z = (v.z - mean) * rstd * w4.z + b4.z;
    o4.w = (v.w - mean) * rstd * w4.w + b4.w;
    *(float4*)(out + (size_t)m * DMODEL + tid * 4) = o4;
}

// ---------------------------------------------------------------------------
// WMMA GEMM: OUT[M,N] = epi( A[M,K] @ W[N,K]^T + bias[N] )
//   ACT==1: exact GELU.  RES==1: OUT = res + gamma (.) val
// 128 threads = 4 waves; tile 128x64 (wave = 32 rows: 2 A-frags x 4 B-frags
// -> 8 WMMA per K-step). Weight tiles streamed via async global->LDS (double
// buffered fp32 staging, ASYNCcnt-tracked) when available; converted to f16.
// ---------------------------------------------------------------------------
__device__ __forceinline__ float gelu_exact(float x) {
    return 0.5f * x * (1.0f + erff(x * 0.70710678118654752f));
}

#define BM 128
#define BN 64
#define BK 32
#define LDA 40   // half-stride: 80B rows, 16B aligned, conflict-staggered

template<int ACT, int RES>
__global__ void wmma_gemm_kernel(const float* __restrict__ A, const float* __restrict__ W,
                                 const float* __restrict__ bias, const float* __restrict__ res,
                                 const float* __restrict__ gamma, float* __restrict__ out,
                                 int M, int N, int K) {
    __shared__ __attribute__((aligned(16))) _Float16 As[BM * LDA];
    __shared__ __attribute__((aligned(16))) _Float16 Ws[BN * LDA];
#if HAVE_ASYNC
    __shared__ __attribute__((aligned(16))) float Wst[2][BN * BK];
#endif
    int tid = threadIdx.x;
    int m0 = blockIdx.y * BM;
    int n0 = blockIdx.x * BN;
    int wave = tid >> 5;
    int lane = tid & 31;
    int hi = lane >> 4;
    int lr = lane & 15;
    v8f acc[2][4] = {};

#if HAVE_ASYNC
    // prologue: kick async load of W tile for k0=0 into staging buffer 0
    // (BN*BK floats = 512 b128 chunks; 4 async instrs per wave)
    for (int i = tid; i < (BN * BK) / 4; i += 128) {
        int row = i >> 3, ch = i & 7;
        int gn = n0 + row;
        int gs = (gn < N) ? gn : (N - 1);
        const float* src = &W[(size_t)gs * K + ch * 4];
        __builtin_amdgcn_global_load_async_to_lds_b128(
            to_global(src), to_lds(&Wst[0][i * 4]), 0, 0);
    }
    int buf = 0;
#endif

    for (int k0 = 0; k0 < K; k0 += BK) {
#if HAVE_ASYNC
        // kick next W tile while current one completes
        if (k0 + BK < K) {
            for (int i = tid; i < (BN * BK) / 4; i += 128) {
                int row = i >> 3, ch = i & 7;
                int gn = n0 + row;
                int gs = (gn < N) ? gn : (N - 1);
                const float* src = &W[(size_t)gs * K + k0 + BK + ch * 4];
                __builtin_amdgcn_global_load_async_to_lds_b128(
                    to_global(src), to_lds(&Wst[buf ^ 1][i * 4]), 0, 0);
            }
            __builtin_amdgcn_s_wait_asynccnt(4);  // current tile's 4 instrs done
        } else {
            __builtin_amdgcn_s_wait_asynccnt(0);
        }
#endif
        // A tile: sync load + convert (activation stream is small / L2-resident)
        for (int i = tid; i < BM * BK; i += 128) {
            int r = i >> 5, c = i & 31;
            int gm = m0 + r;
            float v = (gm < M) ? A[(size_t)gm * K + k0 + c] : 0.f;
            As[r * LDA + c] = (_Float16)v;
        }
#if HAVE_ASYNC
        __syncthreads();  // staging tile visible to whole block
        for (int i = tid; i < BN * BK; i += 128) {
            int r = i >> 5, c = i & 31;
            float v = ((n0 + r) < N) ? Wst[buf][i] : 0.f;
            Ws[r * LDA + c] = (_Float16)v;
        }
        buf ^= 1;
#else
        for (int i = tid; i < BN * BK; i += 128) {
            int r = i >> 5, c = i & 31;
            int gn = n0 + r;
            float v = (gn < N) ? W[(size_t)gn * K + k0 + c] : 0.f;
            Ws[r * LDA + c] = (_Float16)v;
        }
#endif
        __syncthreads();

        U16 a0, a1;
        const _Float16* ap0 = &As[(wave * 32 + lr) * LDA + hi * 8];
        const _Float16* ap1 = &As[(wave * 32 + 16 + lr) * LDA + hi * 8];
        a0.h[0] = *(const v8h*)ap0; a0.h[1] = *(const v8h*)(ap0 + 16);
        a1.h[0] = *(const v8h*)ap1; a1.h[1] = *(const v8h*)(ap1 + 16);
#pragma unroll
        for (int nt = 0; nt < 4; ++nt) {
            U16 bf;
            const _Float16* bp = &Ws[(nt * 16 + lr) * LDA + hi * 8];
            bf.h[0] = *(const v8h*)bp;
            bf.h[1] = *(const v8h*)(bp + 16);
            acc[0][nt] = __builtin_amdgcn_wmma_f32_16x16x32_f16(
                false, a0.v, false, bf.v, (short)0, acc[0][nt], false, false);
            acc[1][nt] = __builtin_amdgcn_wmma_f32_16x16x32_f16(
                false, a1.v, false, bf.v, (short)0, acc[1][nt], false, false);
        }
        __syncthreads();
    }

#pragma unroll
    for (int mi = 0; mi < 2; ++mi) {
#pragma unroll
        for (int nt = 0; nt < 4; ++nt) {
#pragma unroll
            for (int r = 0; r < 8; ++r) {
                int gm = m0 + wave * 32 + mi * 16 + r + 8 * hi;
                int gn = n0 + nt * 16 + lr;
                if (gm < M && gn < N) {
                    float v = acc[mi][nt][r] + bias[gn];
                    if (ACT == 1) v = gelu_exact(v);
                    if (RES) v = res[(size_t)gm * N + gn] + gamma[gn] * v;
                    out[(size_t)gm * N + gn] = v;
                }
            }
        }
    }
}

// ---------------------------------------------------------------------------
// Flash attention with WMMA, one block per (b, head). Q/K staged into LDS with
// fused qk-layernorm (+scale folded into Q); V staged transposed. Hierarchical
// block-causal mask applied as a precomputed per-row key bound.
// ---------------------------------------------------------------------------
__device__ __forceinline__ int key_bound(int q) {
    // token attends keys [0, end[lvl(q)]) ; 0 for padded queries
    if (q >= NTOK) return 0;
    return (q >= 85) ? 341 : (q >= 21) ? 85 : (q >= 5) ? 21 : (q >= 1) ? 5 : 1;
}

__device__ __forceinline__ void ln64_store(float4 v, const float* __restrict__ w,
                                           const float* __restrict__ b,
                                           _Float16* __restrict__ dst, int rl, float mul) {
    float s  = v.x + v.y + v.z + v.w;
    float ss = v.x * v.x + v.y * v.y + v.z * v.z + v.w * v.w;
    for (int o = 1; o < 16; o <<= 1) {
        s  += __shfl_xor(s,  o, 32);
        ss += __shfl_xor(ss, o, 32);
    }
    float mean = s * (1.f / 64.f);
    float var  = ss * (1.f / 64.f) - mean * mean;
    float rstd = rsqrtf(var + 1e-5f);
    int d = rl * 4;
    dst[d + 0] = (_Float16)(((v.x - mean) * rstd * w[d + 0] + b[d + 0]) * mul);
    dst[d + 1] = (_Float16)(((v.y - mean) * rstd * w[d + 1] + b[d + 1]) * mul);
    dst[d + 2] = (_Float16)(((v.z - mean) * rstd * w[d + 2] + b[d + 2]) * mul);
    dst[d + 3] = (_Float16)(((v.w - mean) * rstd * w[d + 3] + b[d + 3]) * mul);
}

#define NPAD 352   // 341 padded to 32

__global__ void attn_kernel(const float* __restrict__ qkv,
                            const float* __restrict__ qnw, const float* __restrict__ qnb,
                            const float* __restrict__ knw, const float* __restrict__ knb,
                            float* __restrict__ outp) {
    extern __shared__ __attribute__((aligned(16))) char smem[];
    _Float16* Qs = (_Float16*)smem;           // [NPAD][64]
    _Float16* Ks = Qs + NPAD * HD;            // [NPAD][64]
    _Float16* Vt = Ks + NPAD * HD;            // [64][NPAD] (transposed)
    _Float16* Ps = Vt + HD * NPAD;            // [4 waves][16][32]

    int bh = blockIdx.x;
    int b = bh >> 4, h = bh & 15;
    int tid = threadIdx.x;
    int wave = tid >> 5, lane = tid & 31, hi = lane >> 4, lr = lane & 15;
    int rid = tid >> 4, rl = tid & 15;

    // ---- stage Q (LN * scale), K (LN), V (transposed) into LDS ----
    for (int n = rid; n < NPAD; n += 8) {
        size_t base = (size_t)(b * NTOK + n) * (3 * DMODEL);
        float4 z; z.x = z.y = z.z = z.w = 0.f;
        float4 q4 = (n < NTOK) ? *(const float4*)(qkv + base + h * HD + rl * 4) : z;
        ln64_store(q4, qnw, qnb, &Qs[n * HD], rl, 0.125f);   // scale = 64^-0.5
        float4 k4 = (n < NTOK) ? *(const float4*)(qkv + base + DMODEL + h * HD + rl * 4) : z;
        ln64_store(k4, knw, knb, &Ks[n * HD], rl, 1.0f);
        float4 v4 = (n < NTOK) ? *(const float4*)(qkv + base + 2 * DMODEL + h * HD + rl * 4) : z;
        int d = rl * 4;
        Vt[(d + 0) * NPAD + n] = (_Float16)v4.x;
        Vt[(d + 1) * NPAD + n] = (_Float16)v4.y;
        Vt[(d + 2) * NPAD + n] = (_Float16)v4.z;
        Vt[(d + 3) * NPAD + n] = (_Float16)v4.w;
    }
    __syncthreads();

    _Float16* pw = Ps + wave * 16 * 32;

    for (int qt = wave; qt < (NPAD / 16); qt += 4) {
        int q0 = qt * 16;
        U16 aq0, aq1;
        const _Float16* qp = &Qs[(q0 + lr) * HD + hi * 8];
        aq0.h[0] = *(const v8h*)qp;        aq0.h[1] = *(const v8h*)(qp + 16);
        aq1.h[0] = *(const v8h*)(qp + 32); aq1.h[1] = *(const v8h*)(qp + 48);

        float m_run[8], l_run[8];
        int   kb[8];                       // per-row allowed-key bound
        v8f o[4] = {};
#pragma unroll
        for (int r = 0; r < 8; ++r) {
            m_run[r] = -3e38f;
            l_run[r] = 0.f;
            kb[r]    = key_bound(q0 + r + 8 * hi);
        }

        for (int kc = 0; kc < NPAD; kc += 32) {
            v8f s0 = {}, s1 = {};
            U16 bk;
            const _Float16* kp0 = &Ks[(kc + lr) * HD + hi * 8];
            bk.h[0] = *(const v8h*)kp0;        bk.h[1] = *(const v8h*)(kp0 + 16);
            s0 = __builtin_amdgcn_wmma_f32_16x16x32_f16(false, aq0.v, false, bk.v, (short)0, s0, false, false);
            bk.h[0] = *(const v8h*)(kp0 + 32); bk.h[1] = *(const v8h*)(kp0 + 48);
            s0 = __builtin_amdgcn_wmma_f32_16x16x32_f16(false, aq1.v, false, bk.v, (short)0, s0, false, false);
            const _Float16* kp1 = &Ks[(kc + 16 + lr) * HD + hi * 8];
            bk.h[0] = *(const v8h*)kp1;        bk.h[1] = *(const v8h*)(kp1 + 16);
            s1 = __builtin_amdgcn_wmma_f32_16x16x32_f16(false, aq0.v, false, bk.v, (short)0, s1, false, false);
            bk.h[0] = *(const v8h*)(kp1 + 32); bk.h[1] = *(const v8h*)(kp1 + 48);
            s1 = __builtin_amdgcn_wmma_f32_16x16x32_f16(false, aq1.v, false, bk.v, (short)0, s1, false, false);

            int k0i = kc + lr, k1i = kc + 16 + lr;
            float p0[8], p1[8];
#pragma unroll
            for (int r = 0; r < 8; ++r) {
                float v0 = (k0i < kb[r]) ? s0[r] : NEGBIG;
                float v1 = (k1i < kb[r]) ? s1[r] : NEGBIG;
                float cm = fmaxf(v0, v1);
                for (int mm = 1; mm < 16; mm <<= 1) cm = fmaxf(cm, __shfl_xor(cm, mm, 32));
                float mnew  = fmaxf(m_run[r], cm);
                float alpha = __expf(m_run[r] - mnew);
                float e0 = __expf(v0 - mnew);
                float e1 = __expf(v1 - mnew);
                float rs = e0 + e1;
                for (int mm = 1; mm < 16; mm <<= 1) rs += __shfl_xor(rs, mm, 32);
                l_run[r] = l_run[r] * alpha + rs;
                m_run[r] = mnew;
                p0[r] = e0; p1[r] = e1;
                o[0][r] *= alpha; o[1][r] *= alpha; o[2][r] *= alpha; o[3][r] *= alpha;
            }

            // re-layout P (C-layout -> A-layout) through per-wave LDS scratch
#pragma unroll
            for (int r = 0; r < 8; ++r) {
                int ml = r + 8 * hi;
                pw[ml * 32 + lr]      = (_Float16)p0[r];
                pw[ml * 32 + 16 + lr] = (_Float16)p1[r];
            }
            U16 ap;
            const _Float16* pp = &pw[lr * 32 + hi * 8];
            ap.h[0] = *(const v8h*)pp;
            ap.h[1] = *(const v8h*)(pp + 16);

#pragma unroll
            for (int nt = 0; nt < 4; ++nt) {
                U16 bv;
                const _Float16* vp = &Vt[(nt * 16 + lr) * NPAD + kc + hi * 8];
                bv.h[0] = *(const v8h*)vp;
                bv.h[1] = *(const v8h*)(vp + 16);
                o[nt] = __builtin_amdgcn_wmma_f32_16x16x32_f16(
                    false, ap.v, false, bv.v, (short)0, o[nt], false, false);
            }
        }

#pragma unroll
        for (int nt = 0; nt < 4; ++nt) {
#pragma unroll
            for (int r = 0; r < 8; ++r) {
                int q = q0 + r + 8 * hi;
                if (q < NTOK) {
                    float val = o[nt][r] / l_run[r];
                    outp[((size_t)(b * NTOK + q)) * DMODEL + h * HD + nt * 16 + lr] = val;
                }
            }
        }
    }
}

// ---------------------------------------------------------------------------
// Host orchestration
// ---------------------------------------------------------------------------
extern "C" void kernel_launch(void* const* d_in, const int* in_sizes, int n_in,
                              void* d_out, int out_size, void* d_ws, size_t ws_size,
                              hipStream_t stream) {
    const float* x      = (const float*)d_in[0];
    const float* qkv_w  = (const float*)d_in[1];
    const float* qkv_b  = (const float*)d_in[2];
    const float* qn_w   = (const float*)d_in[3];
    const float* qn_b   = (const float*)d_in[4];
    const float* kn_w   = (const float*)d_in[5];
    const float* kn_b   = (const float*)d_in[6];
    const float* proj_w = (const float*)d_in[7];
    const float* proj_b = (const float*)d_in[8];
    const float* ln1_w  = (const float*)d_in[9];
    const float* ln1_b  = (const float*)d_in[10];
    const float* ln2_w  = (const float*)d_in[11];
    const float* ln2_b  = (const float*)d_in[12];
    const float* g1     = (const float*)d_in[13];
    const float* g2     = (const float*)d_in[14];
    const float* fc1_w  = (const float*)d_in[15];
    const float* fc1_b  = (const float*)d_in[16];
    const float* fc2_w  = (const float*)d_in[17];
    const float* fc2_b  = (const float*)d_in[18];

    float* h  = (float*)d_out;                       // residual stream [2728,1024]
    float* ws = (float*)d_ws;
    float* y    = ws;                                // [2728,1024]
    float* qkv  = y   + (size_t)MROWS * DMODEL;      // [2728,3072]
    float* att  = qkv + (size_t)MROWS * 3 * DMODEL;  // [2728,1024]
    float* m1   = att + (size_t)MROWS * DMODEL;      // [2728,4096]

    tokens_kernel<<<8 * DMODEL, 256, 0, stream>>>(x, h);

    const int M = MROWS;
    dim3 blk(128);
    dim3 gy((unsigned)((M + BM - 1) / BM));
    size_t attn_lds = (size_t)(NPAD * HD + NPAD * HD + HD * NPAD + 4 * 16 * 32) * sizeof(_Float16);
    (void)hipFuncSetAttribute(reinterpret_cast<const void*>(&attn_kernel),
                              hipFuncAttributeMaxDynamicSharedMemorySize, (int)attn_lds);

    for (int l = 0; l < 8; ++l) {
        ln_kernel<<<M, 256, 0, stream>>>(h, ln1_w + l * DMODEL, ln1_b + l * DMODEL, y);
        wmma_gemm_kernel<0, 0><<<dim3(3 * DMODEL / BN, gy.x), blk, 0, stream>>>(
            y, qkv_w + (size_t)l * 3 * DMODEL * DMODEL, qkv_b + (size_t)l * 3 * DMODEL,
            nullptr, nullptr, qkv, M, 3 * DMODEL, DMODEL);
        attn_kernel<<<8 * NHEADS, 128, attn_lds, stream>>>(
            qkv, qn_w + l * HD, qn_b + l * HD, kn_w + l * HD, kn_b + l * HD, att);
        wmma_gemm_kernel<0, 1><<<dim3(DMODEL / BN, gy.x), blk, 0, stream>>>(
            att, proj_w + (size_t)l * DMODEL * DMODEL, proj_b + (size_t)l * DMODEL,
            h, g1 + (size_t)l * DMODEL, h, M, DMODEL, DMODEL);
        ln_kernel<<<M, 256, 0, stream>>>(h, ln2_w + l * DMODEL, ln2_b + l * DMODEL, y);
        wmma_gemm_kernel<1, 0><<<dim3(FFDIM / BN, gy.x), blk, 0, stream>>>(
            y, fc1_w + (size_t)l * FFDIM * DMODEL, fc1_b + (size_t)l * FFDIM,
            nullptr, nullptr, m1, M, FFDIM, DMODEL);
        wmma_gemm_kernel<0, 1><<<dim3(DMODEL / BN, gy.x), blk, 0, stream>>>(
            m1, fc2_w + (size_t)l * DMODEL * FFDIM, fc2_b + (size_t)l * DMODEL,
            h, g2 + (size_t)l * DMODEL, h, M, DMODEL, FFDIM);
    }
}
